// VFdisc_65738769432780
// MI455X (gfx1250) — compile-verified, matrix operand order checked by reading, not run on hardware
//
#include <hip/hip_runtime.h>

typedef __attribute__((ext_vector_type(16))) __bf16        v16bf;
typedef __attribute__((ext_vector_type(8)))  float         v8f;
typedef __attribute__((ext_vector_type(4)))  unsigned int  v4u;
typedef __attribute__((ext_vector_type(4)))  float         v4f;

#define B_TOT   16384
#define N0v     10
#define N1v     256
#define N2v     256
#define NDv     784
#define TILE_B  32
#define STRIDE  264    // ushorts per row for 256-wide LDS arrays (8-elt pad -> conflict-free b128)
#define SSTR    40     // ushorts per row for 32-wide LDS arrays
#define KCH     8      // 256 / 32

union FragBF { v16bf bf; v4u q[2]; unsigned int u[8]; };

__device__ __forceinline__ unsigned short f2bf(float x) {
  __bf16 h = (__bf16)x;
  return __builtin_bit_cast(unsigned short, h);
}
#if __has_builtin(__builtin_amdgcn_cvt_pk_bf16_f32)
__device__ __forceinline__ unsigned int pk2bf(float lo, float hi) {
  auto r = __builtin_amdgcn_cvt_pk_bf16_f32(lo, hi);   // one v_cvt_pk_bf16_f32
  return __builtin_bit_cast(unsigned int, r);
}
#else
__device__ __forceinline__ unsigned int pk2bf(float lo, float hi) {
  return (unsigned int)f2bf(lo) | ((unsigned int)f2bf(hi) << 16);
}
#endif
__device__ __forceinline__ float bf2f(unsigned short b) {
  return __uint_as_float(((unsigned int)b) << 16);
}
__device__ __forceinline__ float clip01(float x) {
  return fminf(fmaxf(x, 0.0f), 1.0f);
}

__device__ __forceinline__ v8f wmma_bf(FragBF a, FragBF b, v8f c) {
  // D = A(16x32 bf16) * B(32x16 bf16) + C(16x16 f32)
  return __builtin_amdgcn_wmma_f32_16x16x32_bf16(false, a.bf, false, b.bf,
                                                 (short)0, c, false, false);
}

// Fragment: lane L (L<16) holds row/col L, K in [kb, kb+8) and [kb+16, kb+24)
// (caller folds the lane>=16 "+8" K shift into kb). Two 16B LDS loads.
__device__ __forceinline__ FragBF lds_frag(const unsigned short* p) {
  FragBF f;
  f.q[0] = *(const v4u*)(p);
  f.q[1] = *(const v4u*)(p + 16);
  return f;
}

// Same fragment sourced from a global f32 row, packed to bf16.
// full==false: zero the second 8-K group (tail chunk of K=784).
__device__ __forceinline__ FragBF gl_frag(const float* __restrict__ row, int kb, bool full) {
  FragBF f;
  v4f x0 = *(const v4f*)(row + kb);
  v4f x1 = *(const v4f*)(row + kb + 4);
  f.u[0] = pk2bf(x0[0], x0[1]); f.u[1] = pk2bf(x0[2], x0[3]);
  f.u[2] = pk2bf(x1[0], x1[1]); f.u[3] = pk2bf(x1[2], x1[3]);
  if (full) {
    v4f y0 = *(const v4f*)(row + kb + 16);
    v4f y1 = *(const v4f*)(row + kb + 20);
    f.u[4] = pk2bf(y0[0], y0[1]); f.u[5] = pk2bf(y0[2], y0[3]);
    f.u[6] = pk2bf(y1[0], y1[1]); f.u[7] = pk2bf(y1[2], y1[3]);
  } else {
    f.u[4] = 0u; f.u[5] = 0u; f.u[6] = 0u; f.u[7] = 0u;
  }
  return f;
}

// W1 row fragment built once from global (K padded 10 -> 32); T-invariant, lives in VGPRs
__device__ __forceinline__ FragBF w1_frag(const float* __restrict__ W1, int j, int sel) {
  float v[8];
  #pragma unroll
  for (int i = 0; i < 8; ++i) { int k = sel * 8 + i; v[i] = (k < N0v) ? W1[j * N0v + k] : 0.0f; }
  FragBF f;
  f.u[0] = pk2bf(v[0], v[1]); f.u[1] = pk2bf(v[2], v[3]);
  f.u[2] = pk2bf(v[4], v[5]); f.u[3] = pk2bf(v[6], v[7]);
  f.u[4] = 0u; f.u[5] = 0u; f.u[6] = 0u; f.u[7] = 0u;   // K 16..31 of this chunk-group
  return f;
}

// clip + pack 8 accumulator elems (with per-elem bias) into 4 dwords (8 bf16)
__device__ __forceinline__ v4u pack_tile(v8f acc, const float* bias8) {
  float v[8];
  #pragma unroll
  for (int i = 0; i < 8; ++i) v[i] = clip01(acc[i] + bias8[i]);
  v4u r;
  r[0] = pk2bf(v[0], v[1]); r[1] = pk2bf(v[2], v[3]);
  r[2] = pk2bf(v[4], v[5]); r[3] = pk2bf(v[6], v[7]);
  return r;
}

__global__ __launch_bounds__(256, 1)
void vfdisc_fused_kernel(const float* __restrict__ data,
                         const float* __restrict__ s0in,
                         const float* __restrict__ s1in,
                         const float* __restrict__ s2in,
                         const float* __restrict__ W0,
                         const float* __restrict__ b0,
                         const float* __restrict__ W1,
                         const float* __restrict__ W2,
                         const float* __restrict__ b2,
                         const float* __restrict__ W3,
                         const float* __restrict__ W4,
                         const float* __restrict__ b4,
                         const int*   __restrict__ Tptr,
                         float* __restrict__ out)
{
  __shared__ __align__(16) unsigned short lW2[N1v * STRIDE];     // 132.0 KB
  __shared__ __align__(16) unsigned short lW3[N2v * STRIDE];     // 132.0 KB
  __shared__ __align__(16) unsigned short lW0[16 * STRIDE];      //   8.25 KB (rows 10..15 zero)
  __shared__ __align__(16) unsigned short lS1[TILE_B * STRIDE];  //  16.5 KB
  __shared__ __align__(16) unsigned short lS2[TILE_B * STRIDE];  //  16.5 KB
  __shared__ __align__(16) unsigned short lS0[TILE_B * SSTR];    //   2.5 KB  => 315.1 KB total

  const int tid  = threadIdx.x;
  const int wg   = blockIdx.x;
  const int lane = tid & 31;
  const int wave = tid >> 5;
  const int r16  = lane & 15;   // A: neuron row j in tile; B: batch column b in tile
  const int sel  = lane >> 4;   // +8 K offset for upper half-wave

  // ---- stage weights + initial states into LDS as bf16 ----
  for (int i = tid; i < N1v * N2v; i += 256) {
    int j = i >> 8, k = i & 255;
    lW2[j * STRIDE + k] = f2bf(W2[i]);
    lW3[j * STRIDE + k] = f2bf(W3[i]);
  }
  for (int i = tid; i < 16 * 256; i += 256) {
    int j = i >> 8, k = i & 255;
    lW0[j * STRIDE + k] = f2bf(j < N0v ? W0[j * N1v + k] : 0.0f);
  }
  for (int i = tid; i < TILE_B * SSTR; i += 256) lS0[i] = 0;     // pad cols stay 0 forever
  for (int i = tid; i < TILE_B * 256; i += 256) {
    int r = i >> 8, k = i & 255;
    size_t g = (size_t)(wg * TILE_B + r) * 256 + k;
    lS1[r * STRIDE + k] = f2bf(s1in[g]);
    lS2[r * STRIDE + k] = f2bf(s2in[g]);
  }
  __syncthreads();
  for (int i = tid; i < TILE_B * N0v; i += 256) {
    int r = i / N0v, k = i % N0v;
    lS0[r * SSTR + k] = f2bf(s0in[(size_t)(wg * TILE_B + r) * N0v + k]);
  }

  // per-wave: N-tiles t0,t1 of both n1 and n2; M-tiles m=0 (rows 0-15) and m=1 (rows 16-31)
  const int t0 = wave * 2, t1 = wave * 2 + 1;

  // per-VGPR-element biases in D layout (elem i -> neuron j = t*16 + 8*sel + i)
  float b2e0[8], b2e1[8], b0e[8];
  {
    v4f lo, hi;
    lo = *(const v4f*)(b2 + t0 * 16 + sel * 8); hi = *(const v4f*)(b2 + t0 * 16 + sel * 8 + 4);
    #pragma unroll
    for (int i = 0; i < 4; ++i) { b2e0[i] = lo[i]; b2e0[4 + i] = hi[i]; }
    lo = *(const v4f*)(b2 + t1 * 16 + sel * 8); hi = *(const v4f*)(b2 + t1 * 16 + sel * 8 + 4);
    #pragma unroll
    for (int i = 0; i < 4; ++i) { b2e1[i] = lo[i]; b2e1[4 + i] = hi[i]; }
    #pragma unroll
    for (int i = 0; i < 8; ++i) {             // b0 has only 10 elems: guarded scalar loads
      int j = sel * 8 + i;
      b0e[i] = (j < N0v) ? b0[j] : 0.0f;
    }
  }
  const float zero8[8] = {0, 0, 0, 0, 0, 0, 0, 0};

  // T-invariant W1 fragments, kept in VGPRs (removes lW1 + 4 LDS loads/step)
  const FragBF w1f0 = w1_frag(W1, t0 * 16 + r16, sel);
  const FragBF w1f1 = w1_frag(W1, t1 * 16 + r16, sel);

  // ---- drive = data @ W4^T + b4, 4 tiles (t0/t1 x m0/m1), kept in VGPRs (f32) ----
  v8f drv00 = {0,0,0,0,0,0,0,0}, drv01 = {0,0,0,0,0,0,0,0};
  v8f drv10 = {0,0,0,0,0,0,0,0}, drv11 = {0,0,0,0,0,0,0,0};
  {
    const float* ar0 = W4 + (size_t)(t0 * 16 + r16) * NDv;
    const float* ar1 = W4 + (size_t)(t1 * 16 + r16) * NDv;
    const float* br0 = data + (size_t)(wg * TILE_B + r16) * NDv;
    const float* br1 = data + (size_t)(wg * TILE_B + 16 + r16) * NDv;
    #pragma clang loop unroll(disable)
    for (int c = 0; c < 25; ++c) {          // K = 784 -> 24 full chunks + 16-wide tail
      int  kb   = c * 32 + sel * 8;
      bool full = (c < 24);
      FragBF fb0 = gl_frag(br0, kb, full);
      FragBF fb1 = gl_frag(br1, kb, full);
      FragBF fa0 = gl_frag(ar0, kb, full);
      FragBF fa1 = gl_frag(ar1, kb, full);
      drv00 = wmma_bf(fa0, fb0, drv00);
      drv01 = wmma_bf(fa0, fb1, drv01);
      drv10 = wmma_bf(fa1, fb0, drv10);
      drv11 = wmma_bf(fa1, fb1, drv11);
    }
    v4f lo, hi;
    lo = *(const v4f*)(b4 + t0 * 16 + sel * 8); hi = *(const v4f*)(b4 + t0 * 16 + sel * 8 + 4);
    #pragma unroll
    for (int i = 0; i < 4; ++i) { drv00[i] += lo[i]; drv00[4+i] += hi[i];
                                  drv01[i] += lo[i]; drv01[4+i] += hi[i]; }
    lo = *(const v4f*)(b4 + t1 * 16 + sel * 8); hi = *(const v4f*)(b4 + t1 * 16 + sel * 8 + 4);
    #pragma unroll
    for (int i = 0; i < 4; ++i) { drv10[i] += lo[i]; drv10[4+i] += hi[i];
                                  drv11[i] += lo[i]; drv11[4+i] += hi[i]; }
  }

  __syncthreads();

  // ---- T fused recurrence steps, all state in LDS/registers ----
  const int T = Tptr[0];
  #pragma clang loop unroll(disable)
  for (int t = 0; t < T; ++t) {
    v8f a1_00 = {0,0,0,0,0,0,0,0}, a1_01 = {0,0,0,0,0,0,0,0};  // n1 (t0,m0),(t0,m1)
    v8f a1_10 = {0,0,0,0,0,0,0,0}, a1_11 = {0,0,0,0,0,0,0,0};  // n1 (t1,m0),(t1,m1)
    v8f a2_00 = drv00, a2_01 = drv01;                          // n2, C init = drive
    v8f a2_10 = drv10, a2_11 = drv11;
    v8f a0_0  = {0,0,0,0,0,0,0,0}, a0_1 = {0,0,0,0,0,0,0,0};   // n0 (wave 0 only)

    #pragma unroll
    for (int c = 0; c < KCH; ++c) {
      int kb = c * 32 + sel * 8;
      FragBF fs2a = lds_frag(&lS2[r16 * STRIDE + kb]);
      FragBF fs2b = lds_frag(&lS2[(16 + r16) * STRIDE + kb]);
      FragBF fs1a = lds_frag(&lS1[r16 * STRIDE + kb]);
      FragBF fs1b = lds_frag(&lS1[(16 + r16) * STRIDE + kb]);
      FragBF w20  = lds_frag(&lW2[(t0 * 16 + r16) * STRIDE + kb]);
      FragBF w21  = lds_frag(&lW2[(t1 * 16 + r16) * STRIDE + kb]);
      FragBF w30  = lds_frag(&lW3[(t0 * 16 + r16) * STRIDE + kb]);
      FragBF w31  = lds_frag(&lW3[(t1 * 16 + r16) * STRIDE + kb]);
      a1_00 = wmma_bf(w20, fs2a, a1_00);   // A = weights, B = state
      a1_01 = wmma_bf(w20, fs2b, a1_01);
      a1_10 = wmma_bf(w21, fs2a, a1_10);
      a1_11 = wmma_bf(w21, fs2b, a1_11);
      a2_00 = wmma_bf(w30, fs1a, a2_00);
      a2_01 = wmma_bf(w30, fs1b, a2_01);
      a2_10 = wmma_bf(w31, fs1a, a2_10);
      a2_11 = wmma_bf(w31, fs1b, a2_11);
      if (wave == 0) {                     // wave-uniform: EXEC stays all-ones
        FragBF w0f = lds_frag(&lW0[r16 * STRIDE + kb]);
        a0_0 = wmma_bf(w0f, fs1a, a0_0);   // reuse fs1 fragments
        a0_1 = wmma_bf(w0f, fs1b, a0_1);
      }
    }
    { // s0 @ W1^T contribution to n1 (W1 fragments live in VGPRs)
      int kb = sel * 8;
      FragBF fs0a = lds_frag(&lS0[r16 * SSTR + kb]);
      FragBF fs0b = lds_frag(&lS0[(16 + r16) * SSTR + kb]);
      a1_00 = wmma_bf(w1f0, fs0a, a1_00);
      a1_01 = wmma_bf(w1f0, fs0b, a1_01);
      a1_10 = wmma_bf(w1f1, fs0a, a1_10);
      a1_11 = wmma_bf(w1f1, fs0b, a1_11);
    }

    // hard sigmoid + bf16 pack; lane holds batch row b, 8 consecutive neurons
    v4u q1_00 = pack_tile(a1_00, b2e0);
    v4u q1_01 = pack_tile(a1_01, b2e0);
    v4u q1_10 = pack_tile(a1_10, b2e1);
    v4u q1_11 = pack_tile(a1_11, b2e1);
    v4u q2_00 = pack_tile(a2_00, zero8);
    v4u q2_01 = pack_tile(a2_01, zero8);
    v4u q2_10 = pack_tile(a2_10, zero8);
    v4u q2_11 = pack_tile(a2_11, zero8);
    v4u q0_0  = pack_tile(a0_0, b0e);   // cols j>=10 come out exactly 0
    v4u q0_1  = pack_tile(a0_1, b0e);

    __syncthreads();               // everyone done reading old state
    *(v4u*)&lS1[r16        * STRIDE + t0 * 16 + sel * 8] = q1_00;
    *(v4u*)&lS1[(16 + r16) * STRIDE + t0 * 16 + sel * 8] = q1_01;
    *(v4u*)&lS1[r16        * STRIDE + t1 * 16 + sel * 8] = q1_10;
    *(v4u*)&lS1[(16 + r16) * STRIDE + t1 * 16 + sel * 8] = q1_11;
    *(v4u*)&lS2[r16        * STRIDE + t0 * 16 + sel * 8] = q2_00;
    *(v4u*)&lS2[(16 + r16) * STRIDE + t0 * 16 + sel * 8] = q2_01;
    *(v4u*)&lS2[r16        * STRIDE + t1 * 16 + sel * 8] = q2_10;
    *(v4u*)&lS2[(16 + r16) * STRIDE + t1 * 16 + sel * 8] = q2_11;
    if (wave == 0) {
      *(v4u*)&lS0[r16        * SSTR + sel * 8] = q0_0;
      *(v4u*)&lS0[(16 + r16) * SSTR + sel * 8] = q0_1;
    }
    __syncthreads();               // new state visible
  }

  // ---- write outputs: concat(s0, s1, s2) as f32 ----
  float* out0 = out;
  float* out1 = out  + (size_t)B_TOT * N0v;
  float* out2 = out1 + (size_t)B_TOT * N1v;
  for (int i = tid; i < TILE_B * N0v; i += 256) {
    int r = i / N0v, k = i % N0v;
    out0[(size_t)(wg * TILE_B + r) * N0v + k] = bf2f(lS0[r * SSTR + k]);
  }
  for (int i = tid; i < TILE_B * 256; i += 256) {
    int r = i >> 8, k = i & 255;
    size_t g = (size_t)(wg * TILE_B + r) * 256 + k;
    out1[g] = bf2f(lS1[r * STRIDE + k]);
    out2[g] = bf2f(lS2[r * STRIDE + k]);
  }
}

extern "C" void kernel_launch(void* const* d_in, const int* in_sizes, int n_in,
                              void* d_out, int out_size, void* d_ws, size_t ws_size,
                              hipStream_t stream) {
  (void)in_sizes; (void)n_in; (void)out_size; (void)d_ws; (void)ws_size;
  const float* data = (const float*)d_in[0];
  const float* s0in = (const float*)d_in[1];
  const float* s1in = (const float*)d_in[2];
  const float* s2in = (const float*)d_in[3];
  const float* W0   = (const float*)d_in[4];
  const float* b0   = (const float*)d_in[5];
  const float* W1   = (const float*)d_in[6];
  const float* W2   = (const float*)d_in[7];
  const float* b2   = (const float*)d_in[8];
  const float* W3   = (const float*)d_in[9];
  const float* W4   = (const float*)d_in[10];
  const float* b4   = (const float*)d_in[11];
  const int*   Tp   = (const int*)d_in[12];
  float*       outp = (float*)d_out;

  dim3 grid(B_TOT / TILE_B);   // 512 workgroups, one 32-row batch tile each
  dim3 block(256);             // 8 wave32s
  vfdisc_fused_kernel<<<grid, block, 0, stream>>>(data, s0in, s1in, s2in,
                                                  W0, b0, W1, W2, b2, W3, W4, b4,
                                                  Tp, outp);
}
